// CausalSelfAttention_10007273799926
// MI455X (gfx1250) — compile-verified, matrix operand order checked by reading, not run on hardware
//
#include <hip/hip_runtime.h>
#include <hip/hip_bf16.h>

typedef __attribute__((ext_vector_type(16))) __bf16 v16bf;
typedef __attribute__((ext_vector_type(8)))  float  v8f;

#define S_LEN   2048
#define D_MODEL 1024
#define TRIPLE  3072
#define NHEAD   16
#define HDIM    64
#define BATCH   4

// ---------- fragment loader: 16x32 bf16 tile, row-major, per-ISA VGPR layout ----------
// lane L: row = L&15, K chunks [off,off+8) and [16+off,16+off+8), off = (L>>4)*8.
__device__ __forceinline__ v16bf load_frag(const __bf16* __restrict__ base, int stride, int lane) {
    int r   = lane & 15;
    int off = (lane >> 4) * 8;
    const __bf16* p = base + (size_t)r * stride;
    union { v16bf v; uint4 q[2]; } u;
    u.q[0] = *(const uint4*)(p + off);
    u.q[1] = *(const uint4*)(p + off + 16);
    return u.v;
}

__device__ __forceinline__ v16bf load_frag_lds(const __bf16* p, int lane) {
    int r   = lane & 15;
    int off = (lane >> 4) * 8;
    union { v16bf v; uint4 q[2]; } u;
    u.q[0] = *(const uint4*)(p + r * 32 + off);
    u.q[1] = *(const uint4*)(p + r * 32 + off + 16);
    return u.v;
}

__device__ __forceinline__ float rowmax16(float t) {
    t = fmaxf(t, __shfl_xor(t, 1, 32));
    t = fmaxf(t, __shfl_xor(t, 2, 32));
    t = fmaxf(t, __shfl_xor(t, 4, 32));
    t = fmaxf(t, __shfl_xor(t, 8, 32));
    return t;
}
__device__ __forceinline__ float rowsum16(float t) {
    t += __shfl_xor(t, 1, 32);
    t += __shfl_xor(t, 2, 32);
    t += __shfl_xor(t, 4, 32);
    t += __shfl_xor(t, 8, 32);
    return t;
}

// ---------- conversion kernels ----------
__global__ void conv_bf16_kernel(const float* __restrict__ in, __bf16* __restrict__ out, int n) {
    int i = blockIdx.x * blockDim.x + threadIdx.x;
    if (i < n) out[i] = (__bf16)in[i];
}

// in: [K,N] row-major f32  ->  out: [N,K] row-major bf16 (i.e. B^T)
__global__ void conv_transpose_kernel(const float* __restrict__ in, __bf16* __restrict__ out,
                                      int K, int N) {
    int i = blockIdx.x * blockDim.x + threadIdx.x;
    if (i >= K * N) return;
    int n = i / K;
    int k = i - n * K;
    out[i] = (__bf16)in[(size_t)k * N + n];
}

// vt[((b*16+h)*64+hd)*2048 + s] = V[b][s][h][hd]  (V = qkv cols [2048,3072))
__global__ void transpose_v_kernel(const __bf16* __restrict__ qkv, __bf16* __restrict__ vt) {
    int i = blockIdx.x * blockDim.x + threadIdx.x;   // 8388608 total
    int s  = i & (S_LEN - 1);
    int hd = (i >> 11) & (HDIM - 1);
    int h  = (i >> 17) & (NHEAD - 1);
    int b  = i >> 21;
    vt[i] = qkv[(size_t)(b * S_LEN + s) * TRIPLE + 2 * D_MODEL + h * HDIM + hd];
}

// ---------- GEMM: C[M,N] = A[M,K] * BT[N,K]^T, bf16 in, f32 accum ----------
// Each wave computes a 32x64 tile: 2 A fragments share every B fragment
// (8 WMMAs per 12 b128 loads per K=32 step).
template <bool BF16OUT>
__global__ __launch_bounds__(256) void gemm_wmma_kernel(const __bf16* __restrict__ A,
                                                        const __bf16* __restrict__ BT,
                                                        void* __restrict__ Cout,
                                                        int M, int N, int K) {
    const int lane = threadIdx.x & 31;
    const int wave = threadIdx.x >> 5;
    const int mt   = blockIdx.y * 8 + wave;       // 32-row tile index
    const int n0   = blockIdx.x * 64;
    if (mt * 32 >= M) return;

    const v8f z = {0.f, 0.f, 0.f, 0.f, 0.f, 0.f, 0.f, 0.f};
    v8f c0[4] = {z, z, z, z};
    v8f c1[4] = {z, z, z, z};

    const __bf16* arow0 = A + (size_t)(mt * 32) * K;
    const __bf16* arow1 = A + (size_t)(mt * 32 + 16) * K;
    for (int kk = 0; kk < K; kk += 32) {
        v16bf a0 = load_frag(arow0 + kk, K, lane);
        v16bf a1 = load_frag(arow1 + kk, K, lane);
#pragma unroll
        for (int nt = 0; nt < 4; ++nt) {
            v16bf b = load_frag(BT + (size_t)(n0 + nt * 16) * K + kk, K, lane);
            c0[nt] = __builtin_amdgcn_wmma_f32_16x16x32_bf16(false, a0, false, b,
                                                             (short)0, c0[nt], false, false);
            c1[nt] = __builtin_amdgcn_wmma_f32_16x16x32_bf16(false, a1, false, b,
                                                             (short)0, c1[nt], false, false);
        }
    }

    const int half = lane >> 4;
    const int col0 = lane & 15;
#pragma unroll
    for (int nt = 0; nt < 4; ++nt) {
#pragma unroll
        for (int v = 0; v < 8; ++v) {
            int row0 = mt * 32 + v + half * 8;
            int row1 = row0 + 16;
            int col  = n0 + nt * 16 + col0;
            if (BF16OUT) {
                ((__bf16*)Cout)[(size_t)row0 * N + col] = (__bf16)c0[nt][v];
                ((__bf16*)Cout)[(size_t)row1 * N + col] = (__bf16)c1[nt][v];
            } else {
                ((float*)Cout)[(size_t)row0 * N + col] = c0[nt][v];
                ((float*)Cout)[(size_t)row1 * N + col] = c1[nt][v];
            }
        }
    }
}

// ---------- causal flash attention: one wave per (b, h, 16-query tile) ----------
__global__ __launch_bounds__(256) void flash_attn_kernel(const __bf16* __restrict__ qkv,
                                                         const __bf16* __restrict__ vt,
                                                         __bf16* __restrict__ y) {
    __shared__ __bf16 sP[8 * 512];    // 16x32 bf16 P-tile per wave
    const int lane = threadIdx.x & 31;
    const int wave = threadIdx.x >> 5;
    const int wid  = blockIdx.x * 8 + wave;
    const int qt   = wid & 127;
    const int h    = (wid >> 7) & (NHEAD - 1);
    const int b    = wid >> 11;
    const int r    = lane & 15;
    const int half = lane >> 4;

    // Q fragments (held in registers for whole key loop): HD=64 -> two K=32 chunks
    const __bf16* qrow = qkv + (size_t)(b * S_LEN + qt * 16) * TRIPLE + h * HDIM;
    v16bf aQ0 = load_frag(qrow,      TRIPLE, lane);
    v16bf aQ1 = load_frag(qrow + 32, TRIPLE, lane);

    const v8f z = {0.f, 0.f, 0.f, 0.f, 0.f, 0.f, 0.f, 0.f};
    v8f o[4] = {z, z, z, z};
    float rm[8], rl[8];
#pragma unroll
    for (int v = 0; v < 8; ++v) { rm[v] = -1e30f; rl[v] = 0.f; }

    __bf16* pbuf = sP + wave * 512;
    const int kend = qt * 16 + 15;

    for (int kb = 0; kb <= kend; kb += 32) {
        // S = Q K^T for a 16x32 key block (two 16x16 tiles)
        v8f s[2];
#pragma unroll
        for (int nt = 0; nt < 2; ++nt) {
            const __bf16* krow = qkv + (size_t)(b * S_LEN + kb + nt * 16) * TRIPLE
                                 + D_MODEL + h * HDIM;
            v16bf bk0 = load_frag(krow,      TRIPLE, lane);
            v16bf bk1 = load_frag(krow + 32, TRIPLE, lane);
            v8f acc = __builtin_amdgcn_wmma_f32_16x16x32_bf16(false, aQ0, false, bk0,
                                                              (short)0, z, false, false);
            acc = __builtin_amdgcn_wmma_f32_16x16x32_bf16(false, aQ1, false, bk1,
                                                          (short)0, acc, false, false);
            s[nt] = acc;
        }

        // scale + causal mask
#pragma unroll
        for (int nt = 0; nt < 2; ++nt) {
#pragma unroll
            for (int v = 0; v < 8; ++v) {
                float sv = s[nt][v] * 0.125f;      // 1/sqrt(64)
                int qg = qt * 16 + v + half * 8;
                int kg = kb + nt * 16 + r;
                s[nt][v] = (kg > qg) ? -1e30f : sv;
            }
        }

        // online softmax + stage P into LDS (C-layout -> row-major 16x32)
#pragma unroll
        for (int v = 0; v < 8; ++v) {
            float t = rowmax16(fmaxf(s[0][v], s[1][v]));
            float mn = fmaxf(rm[v], t);
            float al = __expf(rm[v] - mn);
            rm[v] = mn;
            rl[v] *= al;
#pragma unroll
            for (int nt = 0; nt < 4; ++nt) o[nt][v] *= al;
            float e0 = __expf(s[0][v] - mn);
            float e1 = __expf(s[1][v] - mn);
            rl[v] += rowsum16(e0 + e1);
            int m = v + half * 8;
            pbuf[m * 32 + r]      = (__bf16)e0;
            pbuf[m * 32 + 16 + r] = (__bf16)e1;
        }
        asm volatile("s_wait_dscnt 0" ::: "memory");   // cross-lane LDS RAW within wave

        // O += P V  (V^T is contiguous along s via vt)
        v16bf aP = load_frag_lds(pbuf, lane);
#pragma unroll
        for (int nt = 0; nt < 4; ++nt) {
            const __bf16* vrow = vt + ((size_t)((b * NHEAD + h) * HDIM + nt * 16)) * S_LEN + kb;
            v16bf bv = load_frag(vrow, S_LEN, lane);
            o[nt] = __builtin_amdgcn_wmma_f32_16x16x32_bf16(false, aP, false, bv,
                                                            (short)0, o[nt], false, false);
        }
    }

    // normalize + write y (bf16, [B*S, D])
#pragma unroll
    for (int v = 0; v < 8; ++v) {
        float inv = 1.0f / rl[v];
        int row = qt * 16 + v + half * 8;
#pragma unroll
        for (int nt = 0; nt < 4; ++nt) {
            int col = h * HDIM + nt * 16 + r;
            y[(size_t)(b * S_LEN + row) * D_MODEL + col] = (__bf16)(o[nt][v] * inv);
        }
    }
}

extern "C" void kernel_launch(void* const* d_in, const int* in_sizes, int n_in,
                              void* d_out, int out_size, void* d_ws, size_t ws_size,
                              hipStream_t stream) {
    const float* x      = (const float*)d_in[0];   // [4,2048,1024]
    const float* w_qkv  = (const float*)d_in[1];   // [1024,3072]
    const float* w_proj = (const float*)d_in[2];   // [1024,1024]
    float* out = (float*)d_out;                    // [4,2048,1024]

    const int M  = BATCH * S_LEN;                  // 8192
    char* ws = (char*)d_ws;
    __bf16* x_bf    = (__bf16*)(ws);                          // 16 MB
    __bf16* wqkvT   = (__bf16*)(ws + 16777216);               //  6 MB
    __bf16* wprojT  = (__bf16*)(ws + 23068672);               //  2 MB
    __bf16* qkv     = (__bf16*)(ws + 25165824);               // 48 MB
    __bf16* vt      = (__bf16*)(ws + 75497472);               // 16 MB
    __bf16* ybuf    = (__bf16*)(ws + 92274688);               // 16 MB

    // 1) conversions
    conv_bf16_kernel<<<(M * D_MODEL + 255) / 256, 256, 0, stream>>>(x, x_bf, M * D_MODEL);
    conv_transpose_kernel<<<(D_MODEL * TRIPLE + 255) / 256, 256, 0, stream>>>(
        w_qkv, wqkvT, D_MODEL, TRIPLE);
    conv_transpose_kernel<<<(D_MODEL * D_MODEL + 255) / 256, 256, 0, stream>>>(
        w_proj, wprojT, D_MODEL, D_MODEL);

    // 2) qkv = x @ w_qkv   (bf16 out)
    {
        dim3 grid(TRIPLE / 64, M / 256);
        gemm_wmma_kernel<true><<<grid, 256, 0, stream>>>(x_bf, wqkvT, (void*)qkv,
                                                         M, TRIPLE, D_MODEL);
    }

    // 3) vt = V^T per (b,h)
    transpose_v_kernel<<<(BATCH * NHEAD * HDIM * S_LEN + 255) / 256, 256, 0, stream>>>(qkv, vt);

    // 4) flash attention -> ybuf
    flash_attn_kernel<<<(BATCH * NHEAD * (S_LEN / 16)) / 8, 256, 0, stream>>>(qkv, vt, ybuf);

    // 5) out = ybuf @ w_proj  (f32 out)
    {
        dim3 grid(D_MODEL / 64, M / 256);
        gemm_wmma_kernel<false><<<grid, 256, 0, stream>>>(ybuf, wprojT, (void*)out,
                                                          M, D_MODEL, D_MODEL);
    }
}
